// FreModule_13735305413212
// MI455X (gfx1250) — compile-verified, matrix operand order checked by reading, not run on hardware
//
#include <hip/hip_runtime.h>
#include <hip/hip_bf16.h>

typedef __attribute__((ext_vector_type(16))) _Float16 v16h;
typedef __attribute__((ext_vector_type(8)))  float    v8f;

#define HW   65536   // 256*256
#define NB   4
#define DIMC 64
#define DWC  128

struct cplx { float x, y; };

__device__ __forceinline__ float sigmoidf_(float v){ return 1.0f/(1.0f+__expf(-v)); }

// ---------------------------------------------------------------------------
// Bilinear upsample x (4,3,128,128) -> xu (4,3,256,256), half-pixel centers
// ---------------------------------------------------------------------------
__global__ void k_upsample(const float* __restrict__ x, float* __restrict__ xu){
  int i = blockIdx.x*256 + threadIdx.x;
  if (i >= NB*3*HW) return;
  int p = i & (HW-1); int bc = i >> 16;
  int ox = p & 255, oy = p >> 8;
  float sx = ox*0.5f - 0.25f;
  float sy = oy*0.5f - 0.25f;
  sx = fminf(fmaxf(sx, 0.0f), 127.0f);
  sy = fminf(fmaxf(sy, 0.0f), 127.0f);
  int x0 = (int)sx, y0 = (int)sy;
  int x1 = min(x0+1, 127), y1 = min(y0+1, 127);
  float fx = sx - (float)x0, fy = sy - (float)y0;
  const float* s = x + bc*(128*128);
  float v00 = s[y0*128+x0], v01 = s[y0*128+x1];
  float v10 = s[y1*128+x0], v11 = s[y1*128+x1];
  xu[i] = v00*(1.f-fx)*(1.f-fy) + v01*fx*(1.f-fy) + v10*(1.f-fx)*fy + v11*fx*fy;
}

// ---------------------------------------------------------------------------
// 3x3 conv, 3 -> 64 channels, pad 1
// ---------------------------------------------------------------------------
__global__ void k_conv3(const float* __restrict__ xu, const float* __restrict__ w,
                        float* __restrict__ out){
  int i = blockIdx.x*256 + threadIdx.x;
  if (i >= NB*DIMC*HW) return;
  int p = i & (HW-1); int bco = i >> 16; int co = bco & 63; int b = bco >> 6;
  int ox = p & 255, oy = p >> 8;
  float acc = 0.f;
  for (int ci = 0; ci < 3; ++ci){
    const float* s  = xu + (b*3 + ci)*HW;
    const float* wk = w + (co*3 + ci)*9;
    #pragma unroll
    for (int ky = 0; ky < 3; ++ky){ int yy = oy+ky-1; if ((unsigned)yy >= 256u) continue;
      #pragma unroll
      for (int kx = 0; kx < 3; ++kx){ int xx = ox+kx-1; if ((unsigned)xx >= 256u) continue;
        acc += s[yy*256+xx]*wk[ky*3+kx]; } }
  }
  out[i] = acc;
}

// ---------------------------------------------------------------------------
// Per-plane mean (and optional max) over H*W.  grid = B*C, block = 256.
// ---------------------------------------------------------------------------
__global__ void k_reduce_plane(const float* __restrict__ in, float* __restrict__ meanOut,
                               float* __restrict__ maxOut, int hw){
  __shared__ float ssum[256];
  __shared__ float smax[256];
  long base = (long)blockIdx.x * hw;
  float s = 0.f, m = -1e30f;
  for (int i = threadIdx.x; i < hw; i += 256){
    float v = in[base+i]; s += v; m = fmaxf(m, v);
  }
  ssum[threadIdx.x] = s; smax[threadIdx.x] = m;
  __syncthreads();
  for (int o = 128; o; o >>= 1){
    if (threadIdx.x < o){
      ssum[threadIdx.x] += ssum[threadIdx.x+o];
      smax[threadIdx.x] = fmaxf(smax[threadIdx.x], smax[threadIdx.x+o]);
    }
    __syncthreads();
  }
  if (threadIdx.x == 0){
    meanOut[blockIdx.x] = ssum[0]/(float)hw;
    if (maxOut) maxOut[blockIdx.x] = smax[0];
  }
}

// ---------------------------------------------------------------------------
// Threshold MLP: mean(4,64) -> gelu(64->8) -> sigmoid(8->2) -> floor(2*s) ints
// ---------------------------------------------------------------------------
__global__ void k_thr(const float* __restrict__ mean, const float* __restrict__ w1,
                      const float* __restrict__ w2, int* __restrict__ thrPar){
  int b = threadIdx.x;
  if (b >= NB) return;
  float hb[8];
  for (int j = 0; j < 8; ++j){
    float s = 0.f;
    for (int c = 0; c < 64; ++c) s += w1[j*64+c]*mean[b*64+c];
    hb[j] = 0.5f*s*(1.0f + erff(s*0.70710678f));   // exact gelu
  }
  for (int o = 0; o < 2; ++o){
    float z = 0.f;
    for (int j = 0; j < 8; ++j) z += w2[o*8+j]*hb[j];
    thrPar[b*2+o] = (int)floorf(2.0f*sigmoidf_(z));  // H//n == W//n == 2
  }
}

// ---------------------------------------------------------------------------
// Serial 256-point radix-2 FFT in LDS (stride = blockDim to spread banks)
// ---------------------------------------------------------------------------
__device__ void fft256(cplx* d, int st, float dir){
  for (int i = 0; i < 256; ++i){
    int j = __brev((unsigned)i) >> 24;
    if (i < j){ cplx t = d[i*st]; d[i*st] = d[j*st]; d[j*st] = t; }
  }
  for (int len = 2; len <= 256; len <<= 1){
    int hlf = len >> 1;
    float ang = dir*6.28318530718f/(float)len;
    for (int i = 0; i < 256; i += len)
      for (int k = 0; k < hlf; ++k){
        float sn, cs; __sincosf(ang*(float)k, &sn, &cs);
        cplx u = d[(i+k)*st], t = d[(i+k+hlf)*st];
        cplx v = { t.x*cs - t.y*sn, t.x*sn + t.y*cs };
        d[(i+k)*st]     = cplx{ u.x+v.x, u.y+v.y };
        d[(i+k+hlf)*st] = cplx{ u.x-v.x, u.y-v.y };
      }
  }
}

// rows: real -> complex, forward, scale 1/256.  32 threads/block, 64KB dyn LDS
__global__ void k_fft_rows_r2c(const float* __restrict__ in, cplx* __restrict__ out){
  extern __shared__ cplx sm[];
  cplx* d = sm + threadIdx.x;
  int st = blockDim.x;
  long row = (long)blockIdx.x*blockDim.x + threadIdx.x;
  const float* src = in + row*256;
  for (int i = 0; i < 256; ++i) d[i*st] = cplx{ src[i], 0.f };
  fft256(d, st, -1.f);
  cplx* dst = out + row*256;
  for (int i = 0; i < 256; ++i){ cplx z = d[i*st]; dst[i] = cplx{ z.x*(1.f/256.f), z.y*(1.f/256.f) }; }
}

// rows: complex -> |ifft|, inverse, no scale (norm='forward')
__global__ void k_fft_rows_c2abs(const cplx* __restrict__ in, float* __restrict__ out){
  extern __shared__ cplx sm[];
  cplx* d = sm + threadIdx.x;
  int st = blockDim.x;
  long row = (long)blockIdx.x*blockDim.x + threadIdx.x;
  const cplx* src = in + row*256;
  for (int i = 0; i < 256; ++i) d[i*st] = src[i];
  fft256(d, st, +1.f);
  float* dst = out + row*256;
  for (int i = 0; i < 256; ++i){ cplx z = d[i*st]; dst[i] = sqrtf(z.x*z.x + z.y*z.y); }
}

// cols: complex -> complex; maskMode 0=none, 1=low band, 2=high band.
// Mask is the fftshift-composed rectangle: keep u,v with ((u+128)&255) in
// [128-h_,128+h_) and ((v+128)&255) in [128-w_,128+w_).
__global__ void k_fft_cols(const cplx* __restrict__ in, cplx* __restrict__ out,
                           float dir, float scale, int maskMode,
                           const int* __restrict__ thrPar){
  extern __shared__ cplx sm[];
  cplx* d = sm + threadIdx.x;
  int st = blockDim.x;
  int col = blockIdx.x*blockDim.x + threadIdx.x;   // B*C*W columns
  int v   = col & 255;
  int img = col >> 8;                               // b*64 + c
  int b   = img >> 6;
  long base = (long)img*HW + v;
  int h_ = 0, w_ = 0; bool cIn = true;
  if (maskMode){
    h_ = thrPar[b*2]; w_ = thrPar[b*2+1];
    int cc = (v+128) & 255;
    cIn = (cc >= 128-w_) && (cc < 128+w_);
  }
  for (int u = 0; u < 256; ++u){
    cplx z = in[base + (long)u*256];
    if (maskMode){
      int rr = (u+128) & 255;
      bool m = cIn && (rr >= 128-h_) && (rr < 128+h_);
      float f = (maskMode == 1) ? (m ? 1.f : 0.f) : (m ? 0.f : 1.f);
      z.x *= f; z.y *= f;
    }
    d[u*st] = z;
  }
  fft256(d, st, dir);
  for (int u = 0; u < 256; ++u){
    cplx z = d[u*st];
    out[base + (long)u*256] = cplx{ z.x*scale, z.y*scale };
  }
}

// ---------------------------------------------------------------------------
// Helper: fragment slot for a K-offset within a 32-wide chunk, per ISA
// 16-bit A/B VGPR layout (lane = idx + 16*hl; vgpr slots pack K pairs).
// ---------------------------------------------------------------------------
__device__ __forceinline__ void frag_map(int kk2, int& ahl, int& slot){
  ahl = (kk2 >> 3) & 1;                    // which lane half carries this K
  int rem = kk2 - ahl*8;                   // {0..7, 16..23}
  int ii  = ((rem >= 16) ? 4 : 0) + ((rem & 7) >> 1);
  slot = 2*ii + (kk2 & 1);                 // element index inside v16h
}

// ---------------------------------------------------------------------------
// 1x1 conv as WMMA GEMM: out[co,p] = sum_ci W[co,ci]*act[ci,p]
// Block = 16 out-ch x 128 px (8 waves x 16 px).  K in chunks of 32:
//   cooperative float4-coalesced load of 32x128 activations, f32->f16,
//   scattered into fragment-major LDS; then 2 LDS b128 reads + v_wmma each.
// Compile-time: CIN (full K unroll), CONCAT (two inputs), SCALE (per-channel
// input scale), BIAS (per-out-channel bias, staged in LDS),
// EPI (out = d*p1[co] + resid*p2[co], p1/p2 staged in LDS).
// ---------------------------------------------------------------------------
template<int CIN, bool CONCAT, bool SCALE, bool BIAS, bool EPI>
__global__ __launch_bounds__(256) void k_conv1x1_wmma(
    const float* __restrict__ in0, int cin0,
    const float* __restrict__ in1, int cin1,
    const float* __restrict__ inScale,
    const float* __restrict__ Wt, const float* __restrict__ bias,
    const float* __restrict__ p1, const float* __restrict__ p2,
    const float* __restrict__ resid,
    float* __restrict__ out, int Cout)
{
  constexpr int KC = CIN/32;
  __shared__ __align__(32) _Float16 Alds[KC*32*16];  // 16 rows x CIN, fragment-major
  __shared__ __align__(32) _Float16 Blds[8*32*16];   // 128 px x 32 k, fragment-major
  __shared__ float Bias[16];                         // per-block out-channel params
  __shared__ float P1[16], P2[16];

  const int tid  = threadIdx.x;
  const int lane = tid & 31;
  const int wv   = tid >> 5;
  const int coB  = blockIdx.x * 16;
  const int g    = blockIdx.y * 128;        // block pixel base over B*HW
  const int b    = g >> 16;                 // HW == 65536; 128 | HW so b uniform
  const int p    = g & (HW-1);
  const int n    = lane & 15;
  const int hl   = lane >> 4;

  // ---- stage this block's 16 weight rows into fragment-major LDS ----
  for (int idx = tid; idx < 16*CIN; idx += 256){
    int m  = idx / CIN;
    int ci = idx - m*CIN;
    int kc = ci >> 5;
    int ahl, slot; frag_map(ci & 31, ahl, slot);
    Alds[(kc*32 + (m + 16*ahl))*16 + slot] = (_Float16)Wt[(coB+m)*CIN + ci];
  }
  // ---- stage per-out-channel epilogue params (branch-free epilogue) ----
  if (tid < 16){
    if (BIAS) Bias[tid] = bias[coB + tid];
    if (EPI){ P1[tid] = p1[coB + tid]; P2[tid] = p2[coB + tid]; }
  }

  v8f acc = {};
  #pragma unroll
  for (int kc = 0; kc < KC; ++kc){
    __syncthreads();
    // ---- stage 32(k) x 128(px) activations ----
    #pragma unroll
    for (int r = 0; r < 4; ++r){
      int idx = (r*256 + tid) * 4;          // element index (ci-major, px fastest)
      int px  = idx & 127;
      int cl  = idx >> 7;                   // 0..31
      int ci  = kc*32 + cl;
      const float* srcp;
      if (CONCAT && ci >= cin0) srcp = in1 + (size_t)(b*cin1 + (ci - cin0))*HW;
      else                      srcp = in0 + (size_t)(b*cin0 + ci)*HW;
      float4 v4 = *reinterpret_cast<const float4*>(srcp + p + px);
      if (KC > 1 && kc + 1 < KC)
        __builtin_prefetch(srcp + 32*HW + p + px, 0, 0);   // global_prefetch_b8
      float scl = 1.0f;
      if (SCALE) scl = inScale[b*CIN + ci];
      int ahl, slot; frag_map(cl, ahl, slot);
      float vv[4] = { v4.x, v4.y, v4.z, v4.w };
      #pragma unroll
      for (int q = 0; q < 4; ++q){
        int pxq = px + q;
        Blds[(((pxq >> 4)*32) + ((pxq & 15) + 16*ahl))*16 + slot] = (_Float16)(vv[q]*scl);
      }
    }
    __syncthreads();
    v16h av = *reinterpret_cast<const v16h*>(&Alds[(kc*32 + lane)*16]);
    v16h bv = *reinterpret_cast<const v16h*>(&Blds[(wv*32 + lane)*16]);
    acc = __builtin_amdgcn_wmma_f32_16x16x32_f16(false, av, false, bv,
                                                 (short)0, acc, false, false);
  }

  #pragma unroll
  for (int j = 0; j < 8; ++j){
    int jc = j + 8*hl;                       // ISA 32-bit C/D layout
    int co = coB + jc;
    float v = acc[j];
    if (BIAS) v += Bias[jc];
    long oi = ((long)(b*Cout) + co)*HW + p + wv*16 + n;
    if (EPI) v = v*P1[jc] + resid[oi]*P2[jc];
    out[oi] = v;
  }
}

// ---------------------------------------------------------------------------
// Depthwise 3x3 (128 ch) + bias + SimpleGate (ch c * ch c+64) fused
// ---------------------------------------------------------------------------
__global__ void k_dwgate(const float* __restrict__ t, const float* __restrict__ w,
                         const float* __restrict__ bias, float* __restrict__ out){
  int i = blockIdx.x*256 + threadIdx.x;
  if (i >= NB*DIMC*HW) return;
  int p = i & (HW-1); int bc = i >> 16; int c = bc & 63; int b = bc >> 6;
  int ox = p & 255, oy = p >> 8;
  float a  = bias[c];
  float g2 = bias[c+64];
  const float* s1 = t + (b*DWC + c)*HW;
  const float* s2 = t + (b*DWC + c + 64)*HW;
  const float* w1 = w + c*9;
  const float* w2 = w + (c+64)*9;
  #pragma unroll
  for (int ky = 0; ky < 3; ++ky){ int yy = oy+ky-1; if ((unsigned)yy >= 256u) continue;
    #pragma unroll
    for (int kx = 0; kx < 3; ++kx){ int xx = ox+kx-1; if ((unsigned)xx >= 256u) continue;
      int q = yy*256+xx;
      a  += s1[q]*w1[ky*3+kx];
      g2 += s2[q]*w2[ky*3+kx];
    } }
  out[i] = a*g2;
}

// Channel-attention vector: svec[b,co] = bias[co] + sum_ci W[co,ci]*pooled[b,ci]
__global__ void k_scavec(const float* __restrict__ pooled, const float* __restrict__ w,
                         const float* __restrict__ bias, float* __restrict__ svec){
  int i = threadIdx.x;              // 256 = 4*64
  int b = i >> 6, co = i & 63;
  float s = bias[co];
  for (int ci = 0; ci < 64; ++ci) s += w[co*64+ci]*pooled[b*64+ci];
  svec[i] = s;
}

// Per-pixel channel max & mean over 64 channels
__global__ void k_chanstats(const float* __restrict__ h, float* __restrict__ smax,
                            float* __restrict__ smean){
  int i = blockIdx.x*256 + threadIdx.x;
  if (i >= NB*HW) return;
  int b = i >> 16; int p = i & (HW-1);
  float mx = -1e30f, s = 0.f;
  for (int c = 0; c < 64; ++c){
    float v = h[(b*64 + c)*HW + p];
    mx = fmaxf(mx, v); s += v;
  }
  smax[i] = mx; smean[i] = s*(1.f/64.f);
}

// 7x7 spatial gate conv over [smax, smean], pad 3, sigmoid
__global__ void k_spatial(const float* __restrict__ smax, const float* __restrict__ smean,
                          const float* __restrict__ sg, float* __restrict__ sw){
  int i = blockIdx.x*256 + threadIdx.x;
  if (i >= NB*HW) return;
  int b = i >> 16; int p = i & (HW-1);
  int ox = p & 255, oy = p >> 8;
  float acc = 0.f;
  for (int ky = 0; ky < 7; ++ky){ int yy = oy+ky-3; if ((unsigned)yy >= 256u) continue;
    for (int kx = 0; kx < 7; ++kx){ int xx = ox+kx-3; if ((unsigned)xx >= 256u) continue;
      int q = b*HW + yy*256+xx;
      acc += smax[q]*sg[ky*7+kx] + smean[q]*sg[49 + ky*7+kx];
    } }
  sw[i] = sigmoidf_(acc);
}

// channel gate: sigmoid(W2*relu(W1*avg) + W2*relu(W1*max)).  grid=B, block=64
__global__ void k_changate(const float* __restrict__ avg, const float* __restrict__ mx,
                           const float* __restrict__ w1, const float* __restrict__ w2,
                           float* __restrict__ cw){
  int b = blockIdx.x;
  __shared__ float h1[4], h2[4];
  if (threadIdx.x < 4){
    float s1 = 0.f, s2 = 0.f;
    for (int c = 0; c < 64; ++c){
      s1 += w1[threadIdx.x*64+c]*avg[b*64+c];
      s2 += w1[threadIdx.x*64+c]*mx[b*64+c];
    }
    h1[threadIdx.x] = fmaxf(s1, 0.f);
    h2[threadIdx.x] = fmaxf(s2, 0.f);
  }
  __syncthreads();
  int c = threadIdx.x;
  float z = 0.f;
  for (int j = 0; j < 4; ++j) z += w2[c*4+j]*(h1[j]+h2[j]);
  cw[b*64+c] = sigmoidf_(z);
}

// comb = low*sw[b,p] + high*cw[b,c]
__global__ void k_combine(const float* __restrict__ lo, const float* __restrict__ hi,
                          const float* __restrict__ sw, const float* __restrict__ cwv,
                          float* __restrict__ out){
  int i = blockIdx.x*256 + threadIdx.x;
  if (i >= NB*DIMC*HW) return;
  int p = i & (HW-1); int bc = i >> 16; int c = bc & 63; int b = bc >> 6;
  out[i] = lo[i]*sw[b*HW+p] + hi[i]*cwv[b*64+c];
}

// ---------------------------------------------------------------------------
extern "C" void kernel_launch(void* const* d_in, const int* in_sizes, int n_in,
                              void* d_out, int out_size, void* d_ws, size_t ws_size,
                              hipStream_t stream){
  (void)in_sizes; (void)n_in; (void)out_size; (void)ws_size;
  const float* x     = (const float*)d_in[0];
  const float* y     = (const float*)d_in[1];
  const float* conv1w= (const float*)d_in[2];
  const float* ratew1= (const float*)d_in[3];
  const float* ratew2= (const float*)d_in[4];
  const float* c1w   = (const float*)d_in[5];
  const float* c1b   = (const float*)d_in[6];
  const float* c2w   = (const float*)d_in[7];
  const float* c2b   = (const float*)d_in[8];
  const float* scaw  = (const float*)d_in[9];
  const float* scab  = (const float*)d_in[10];
  const float* c3w   = (const float*)d_in[11];
  const float* c3b   = (const float*)d_in[12];
  const float* mixw  = (const float*)d_in[13];
  const float* sgw   = (const float*)d_in[14];
  const float* cgw1  = (const float*)d_in[15];
  const float* cgw2  = (const float*)d_in[16];
  const float* projw = (const float*)d_in[17];
  const float* projb = (const float*)d_in[18];
  const float* para1 = (const float*)d_in[19];
  const float* para2 = (const float*)d_in[20];
  float* outp = (float*)d_out;

  // ---- workspace layout (bump allocator with aggressive reuse) ----
  char* ws = (char*)d_ws;
  size_t o = 0;
  auto alloc = [&](size_t bytes){ size_t r = o; o += (bytes + 255) & ~(size_t)255; return r; };
  const size_t REAL = (size_t)NB*DIMC*HW*4;     // 67MB
  size_t XUo = alloc((size_t)NB*3*HW*4);
  size_t XCo = alloc(REAL);        // xc -> LOW -> MIXL
  size_t CAo = alloc(REAL*2);      // complex row-fft / T (B,128,HW) / COMB
  size_t CBo = alloc(REAL*2);      // complex FZ / [GATED | SCAT]
  size_t HIo = alloc(REAL);        // HIGH -> MIXH
  size_t SYo = alloc(REAL);        // sca(y) output
  size_t AGo = alloc(REAL);        // agg
  size_t MEo = alloc(256*4);
  size_t THo = alloc(256);
  size_t POo = alloc(256*4);
  size_t SVo = alloc(256*4);
  size_t AVo = alloc(256*4);
  size_t MXo = alloc(256*4);
  size_t CWo = alloc(256*4);
  size_t SMo = alloc((size_t)NB*HW*4);
  size_t SNo = alloc((size_t)NB*HW*4);
  size_t SWo = alloc((size_t)NB*HW*4);

  float* XU    = (float*)(ws+XUo);
  float* XC    = (float*)(ws+XCo);     // also LOW / MIXL
  cplx*  CA    = (cplx*) (ws+CAo);
  cplx*  CB    = (cplx*) (ws+CBo);
  float* T     = (float*)(ws+CAo);     // (B,128,HW) sca expand, aliases CA
  float* COMB  = (float*)(ws+CAo);     // aliases CA after T is dead
  float* GATED = (float*)(ws+CBo);
  float* SCAT  = (float*)(ws+CBo) + (size_t)NB*DIMC*HW;
  float* HIGH  = (float*)(ws+HIo);     // also MIXH
  float* SCAY  = (float*)(ws+SYo);
  float* AGG   = (float*)(ws+AGo);
  float* MEANS = (float*)(ws+MEo);
  int*   THP   = (int*)  (ws+THo);
  float* POOL  = (float*)(ws+POo);
  float* SVEC  = (float*)(ws+SVo);
  float* AVGP  = (float*)(ws+AVo);
  float* MAXP  = (float*)(ws+MXo);
  float* CWv   = (float*)(ws+CWo);
  float* SMAX  = (float*)(ws+SMo);
  float* SMEAN = (float*)(ws+SNo);
  float* SW    = (float*)(ws+SWo);

  const int EB = 65536;                 // elementwise blocks for B*64*HW
  const int PB = 1024;                  // blocks for B*HW
  const dim3 blk(256);
  const dim3 gemmGrid8(8, 2048);        // Cout=128
  const dim3 gemmGrid4(4, 2048);        // Cout=64
  const int FFT_T = 32;                 // fft threads/block
  const size_t FFT_LDS = (size_t)FFT_T*256*sizeof(cplx);   // 64KB
  const int FFT_B = (NB*DIMC*256)/FFT_T;                    // 2048 blocks

  // full SCA block: src(B,64,HW) -> dst(B,64,HW), weight set i
  auto sca_run = [&](const float* src, int i, float* dst){
    k_conv1x1_wmma<64,false,false,true,false><<<gemmGrid8, blk, 0, stream>>>(
        src, 64, nullptr, 0, nullptr, c1w + (size_t)i*DWC*DIMC, c1b + i*DWC,
        nullptr, nullptr, nullptr, T, DWC);
    k_dwgate<<<EB, blk, 0, stream>>>(T, c2w + (size_t)i*DWC*9, c2b + i*DWC, GATED);
    k_reduce_plane<<<NB*DIMC, blk, 0, stream>>>(GATED, POOL, nullptr, HW);
    k_scavec<<<1, blk, 0, stream>>>(POOL, scaw + (size_t)i*DIMC*DIMC, scab + i*DIMC, SVEC);
    k_conv1x1_wmma<64,false,true,true,false><<<gemmGrid4, blk, 0, stream>>>(
        GATED, 64, nullptr, 0, SVEC, c3w + (size_t)i*DIMC*DIMC, c3b + i*DIMC,
        nullptr, nullptr, nullptr, dst, DIMC);
  };

  // ---- stem: upsample + conv3x3 + threshold MLP ----
  k_upsample<<<(NB*3*HW+255)/256, blk, 0, stream>>>(x, XU);
  k_conv3<<<EB, blk, 0, stream>>>(XU, conv1w, XC);
  k_reduce_plane<<<NB*DIMC, blk, 0, stream>>>(XC, MEANS, nullptr, HW);
  k_thr<<<1, 32, 0, stream>>>(MEANS, ratew1, ratew2, THP);

  // ---- FFT split: fft2 (norm=forward), masked ifft2 -> |high|, |low| ----
  k_fft_rows_r2c<<<FFT_B, FFT_T, FFT_LDS, stream>>>(XC, CA);
  k_fft_cols<<<FFT_B, FFT_T, FFT_LDS, stream>>>(CA, CB, -1.f, 1.f/256.f, 0, THP);  // CB = FZ
  k_fft_cols<<<FFT_B, FFT_T, FFT_LDS, stream>>>(CB, CA, +1.f, 1.f, 2, THP);        // high band
  k_fft_rows_c2abs<<<FFT_B, FFT_T, FFT_LDS, stream>>>(CA, HIGH);
  k_fft_cols<<<FFT_B, FFT_T, FFT_LDS, stream>>>(CB, CA, +1.f, 1.f, 1, THP);        // low band
  k_fft_rows_c2abs<<<FFT_B, FFT_T, FFT_LDS, stream>>>(CA, XC);                     // LOW -> XC

  // ---- refinement: sca + mix for high and low ----
  sca_run(HIGH, 0, SCAT);
  k_conv1x1_wmma<128,true,false,false,false><<<gemmGrid4, blk, 0, stream>>>(
      SCAT, 64, y, 64, nullptr, mixw + 0*DIMC*DWC, nullptr,
      nullptr, nullptr, nullptr, HIGH, DIMC);                                      // MIXH
  sca_run(XC, 1, SCAT);
  k_conv1x1_wmma<128,true,false,false,false><<<gemmGrid4, blk, 0, stream>>>(
      SCAT, 64, y, 64, nullptr, mixw + 1*DIMC*DWC, nullptr,
      nullptr, nullptr, nullptr, XC, DIMC);                                        // MIXL

  // ---- FreRefine ----
  k_chanstats<<<PB, blk, 0, stream>>>(HIGH, SMAX, SMEAN);
  k_spatial<<<PB, blk, 0, stream>>>(SMAX, SMEAN, sgw, SW);
  k_reduce_plane<<<NB*DIMC, blk, 0, stream>>>(XC, AVGP, MAXP, HW);
  k_changate<<<NB, 64, 0, stream>>>(AVGP, MAXP, cgw1, cgw2, CWv);
  k_combine<<<EB, blk, 0, stream>>>(XC, HIGH, SW, CWv, COMB);
  k_conv1x1_wmma<64,false,false,true,false><<<gemmGrid4, blk, 0, stream>>>(
      COMB, 64, nullptr, 0, nullptr, projw, projb,
      nullptr, nullptr, nullptr, AGG, DIMC);

  // ---- residual branch sca(y) + final mix with para1/para2 epilogue ----
  sca_run(y, 2, SCAY);
  k_conv1x1_wmma<128,true,false,false,true><<<gemmGrid4, blk, 0, stream>>>(
      AGG, 64, SCAY, 64, nullptr, mixw + 2*DIMC*DWC, nullptr,
      para1, para2, y, outp, DIMC);
}
